// LSTM_28913719836976
// MI455X (gfx1250) — matrix operand order repaired and checked by reading, not compiled, not run on hardware
//
#include <hip/hip_runtime.h>
#include <math.h>

// Problem dims
constexpr int kB  = 128;   // batch
constexpr int kS  = 1024;  // sequence
constexpr int kD  = 256;   // input dim
constexpr int kH  = 256;   // hidden dim
constexpr int kZK = kD + kH;            // 512 = GEMM K
constexpr int kHS = 64;                 // hidden units per workgroup slice
constexpr int kWC = 4 * kHS;            // 256 W columns per WG (4 gates x 64 units)
constexpr int kNSLICE = kH / kHS;       // 4
constexpr int kBTILES = kB / 16;        // 8
constexpr int kNWG = kNSLICE * kBTILES; // 32 persistent workgroups
constexpr int kBLOCK = 256;             // 8 wave32s
constexpr int kKSTEPS = kZK / 32;       // 16 WMMA K-steps

typedef __attribute__((ext_vector_type(16))) __bf16 v16bf;
typedef __attribute__((ext_vector_type(8)))  float  v8f;

union Frag16 { v16bf v; uint4 q[2]; };

__device__ __forceinline__ float sigmoidf_(float x) {
    return 1.0f / (1.0f + __expf(-x));
}

__global__ __launch_bounds__(kBLOCK, 1)
void lstm_persistent_wmma(const float* __restrict__ x,
                          const float* __restrict__ mask,
                          const float* __restrict__ Wi, const float* __restrict__ bi,
                          const float* __restrict__ Wf, const float* __restrict__ bfp,
                          const float* __restrict__ Wo, const float* __restrict__ bo,
                          const float* __restrict__ Wc, const float* __restrict__ bc,
                          float* __restrict__ out,
                          float* __restrict__ ws) {
    // Small static LDS: z staging + gate tile + biases + lengths (~34 KB)
    __shared__ __bf16 sZ[16 * kZK];   // [row][k]
    __shared__ float  sG[16 * kWC];   // [row][col], col = gate*64 + local_unit
    __shared__ float  sB[kWC];
    __shared__ float  sLen[16];

    const int tid   = threadIdx.x;
    const int wg    = blockIdx.x;
    const int btile = wg / kNSLICE;         // 0..7
    const int nslc  = wg % kNSLICE;         // 0..3
    const int hbase = nslc * kHS;
    const int lane  = tid & 31;
    const int wave  = tid >> 5;             // 0..7 -> output cols [wave*32, wave*32+32)

    float* hbuf = ws;                       // [2][B][H] ping-pong h state
    float* cbuf = ws + 2 * kB * kH;         // [B][H]    c state (private per WG slice)
    unsigned* bar = (unsigned*)(ws + 3 * kB * kH); // bar[0]=count, bar[1]=generation

    const float* Wg[4] = {Wi, Wf, Wo, Wc};
    const float* bg[4] = {bi, bfp, bo, bc};

    {   // biases, column layout col = gate*64 + local_unit
        int gate = tid >> 6, u = hbase + (tid & 63);
        sB[tid] = bg[gate][u];
    }

    // WMMA fragment addressing (wave32; A and B use the same contiguous k order, k = 16*(lane/16) + s)
    const int rowA  = lane & 15;
    const int kselA = (lane >> 4) ? 16 : 0;  // A: lanes 16-31 hold K=16..31
    const int nlo   = lane & 15;
    const int kselB = (lane >> 4) ? 16 : 0;  // B: lanes 16-31 hold K=16..31
    const int col0  = wave * 32;

    // ---- one-time: this wave's 32 W columns -> 32 register-resident B fragments.
    // W column c (= gate*64 + local_unit) is row u of W_gate, contiguous over k,
    // so each lane reads 16 consecutive f32 and converts to bf16 in registers.
    v16bf breg[2 * kKSTEPS];
#pragma unroll
    for (int ks16 = 0; ks16 < kKSTEPS; ++ks16) {
#pragma unroll
        for (int half = 0; half < 2; ++half) {
            int c = col0 + half * 16 + nlo;
            int gate = c >> 6, u = hbase + (c & 63);
            const float* p = Wg[gate] + (size_t)u * kZK + ks16 * 32 + kselB;
            v16bf f;
#pragma unroll
            for (int j = 0; j < 16; ++j) f[j] = (__bf16)p[j];
            breg[ks16 * 2 + half] = f;
        }
    }
    __syncthreads();

    for (int t = 0; t < kS; ++t) {
        const float* hprev = hbuf + (t & 1) * (kB * kH);
        float*       hnext = hbuf + ((t + 1) & 1) * (kB * kH);

        // ---- stage z = [x_t | h_prev] (16 rows x 512) into LDS as bf16
        for (int i = tid; i < 16 * kZK; i += kBLOCK) {
            int row = i >> 9, k = i & (kZK - 1);
            int b = btile * 16 + row;
            float v = (k < kD) ? x[(size_t)b * kS * kD + (size_t)t * kD + k]
                               : hprev[b * kH + (k - kD)];
            sZ[i] = (__bf16)v;
            // prefetch next step's x slab (one request per 256B line)
            if ((k & 63) == 0 && k < kD && (t + 1) < kS)
                __builtin_prefetch(&x[(size_t)b * kS * kD + (size_t)(t + 1) * kD + k], 0, 1);
        }
        __syncthreads();

        // ---- preload the full A tile for this step (16 frags, burst of ds_load_b128)
        Frag16 afr[kKSTEPS];
#pragma unroll
        for (int ks16 = 0; ks16 < kKSTEPS; ++ks16) {
            const __bf16* zp = sZ + rowA * kZK + ks16 * 32 + kselA;
            afr[ks16].q[0] = *(const uint4*)(zp);
            afr[ks16].q[1] = *(const uint4*)(zp + 8);
        }

        // ---- 32 back-to-back WMMAs: A and B both register-resident
        v8f acc0 = {};
        v8f acc1 = {};
#pragma unroll
        for (int ks16 = 0; ks16 < kKSTEPS; ++ks16) {
            acc0 = __builtin_amdgcn_wmma_f32_16x16x32_bf16(false, afr[ks16].v, false,
                                                           breg[2 * ks16],
                                                           (short)0, acc0, false, false);
            acc1 = __builtin_amdgcn_wmma_f32_16x16x32_bf16(false, afr[ks16].v, false,
                                                           breg[2 * ks16 + 1],
                                                           (short)0, acc1, false, false);
        }

        // ---- scatter D (C/D layout: VGPR r -> M=r (lanes 0-15) / M=r+8 (lanes 16-31))
        {
            const int rbase = (lane >> 4) ? 8 : 0;
#pragma unroll
            for (int r = 0; r < 8; ++r) {
                sG[(rbase + r) * kWC + col0 + nlo]      = acc0[r];
                sG[(rbase + r) * kWC + col0 + 16 + nlo] = acc1[r];
            }
        }
        __syncthreads();

        // ---- LSTM pointwise update for this WG's (16 rows x 64 units) slice
        for (int it = tid; it < 16 * kHS; it += kBLOCK) {
            int row = it >> 6, u = it & 63;
            int b = btile * 16 + row;
            int hg = hbase + u;
            float ig = sigmoidf_(sG[row * kWC +       u] + sB[      u]);
            float fg = sigmoidf_(sG[row * kWC +  64 + u] + sB[ 64 + u]);
            float og = sigmoidf_(sG[row * kWC + 128 + u] + sB[128 + u]);
            float cg = tanhf(    sG[row * kWC + 192 + u] + sB[192 + u]);
            float cprev = cbuf[b * kH + hg];
            float cnew = fg * cprev + ig * cg;
            float hnew = tanhf(cnew) * og;
            cbuf[b * kH + hg] = cnew;
            hnext[b * kH + hg] = hnew;
            out[(size_t)b * kS * kH + (size_t)t * kH + hg] = hnew;
        }

        // ---- grid-wide generation barrier: h(t+1) visible before anyone reads it
        __syncthreads();
        if (tid == 0) {
            __threadfence();
            unsigned gen  = atomicAdd(&bar[1], 0u);
            unsigned prev = atomicAdd(&bar[0], 1u);
            if (prev == (unsigned)(kNWG - 1)) {
                atomicExch(&bar[0], 0u);
                __threadfence();
                atomicAdd(&bar[1], 1u);
            } else {
                while (atomicAdd(&bar[1], 0u) == gen) { __builtin_amdgcn_s_sleep(1); }
            }
        }
        __syncthreads();
    }

    // ---- hn = output[b, idx, :] with idx from mask lengths (reads only our own slice)
    if (tid < 16) {
        int b = btile * 16 + tid;
        float s = 0.f;
        for (int t2 = 0; t2 < kS; ++t2) s += mask[(size_t)b * kS + t2];
        sLen[tid] = s;
    }
    __syncthreads();
    for (int it = tid; it < 16 * kHS; it += kBLOCK) {
        int row = it >> 6, u = it & 63;
        int b = btile * 16 + row;
        int idx = (int)sLen[row] - 1;
        if (idx <= 0) idx += 1;
        int hg = hbase + u;
        out[(size_t)kB * kS * kH + (size_t)b * kH + hg] =
            out[(size_t)b * kS * kH + (size_t)idx * kH + hg];
    }
}

__global__ void init_ws(float* ws, int n) {
    int i = blockIdx.x * blockDim.x + threadIdx.x;
    if (i < n) ws[i] = 0.0f;   // zeros h0, c0, and both barrier words
}

extern "C" void kernel_launch(void* const* d_in, const int* in_sizes, int n_in,
                              void* d_out, int out_size, void* d_ws, size_t ws_size,
                              hipStream_t stream) {
    (void)in_sizes; (void)n_in; (void)out_size; (void)ws_size;
    const float* x    = (const float*)d_in[0];
    const float* mask = (const float*)d_in[1];
    // d_in[2]=user_states, d_in[3]=item_states: unused by the executed reference path
    const float* Wi = (const float*)d_in[4];
    const float* bi = (const float*)d_in[5];
    const float* Wf = (const float*)d_in[6];
    const float* bf = (const float*)d_in[7];
    const float* Wo = (const float*)d_in[8];
    const float* bo = (const float*)d_in[9];
    const float* Wc = (const float*)d_in[10];
    const float* bc = (const float*)d_in[11];
    float* out = (float*)d_out;
    float* ws  = (float*)d_ws;

    const int initN = 3 * kB * kH + 2;   // h ping-pong + c state + barrier words
    init_ws<<<(initN + 255) / 256, 256, 0, stream>>>(ws, initN);

    lstm_persistent_wmma<<<kNWG, kBLOCK, 0, stream>>>(
        x, mask, Wi, bi, Wf, bf, Wo, bo, Wc, bc, out, ws);
}
